// GIN_ModelBenX_45792941310041
// MI455X (gfx1250) — matrix operand order repaired and checked by reading, not compiled, not run on hardware
//
#include <hip/hip_runtime.h>
#include <hip/hip_bf16.h>

// ---------------------------------------------------------------------------
// GIN 3-layer forward for MI455X (gfx1250, wave32).
//   agg_i = sum_{(s,d): d=i} x_s   -> one wave per edge, float4/lane,
//                                     HW global_atomic_add_f32 into L2-resident agg
//   h = relu((x+agg) @ W + b)      -> WMMA f32_16x16x32_f16, f32 accumulate
// N=100000, E=1600000, dims 128->128->128->64.
// ---------------------------------------------------------------------------

#define N_NODES 100000
#define N_EDGES 1600000
#define KDIM    128
#define LPAD    136   // padded LDS row stride in halfs (272B -> bank rotation)
#define GROWS   64    // node rows per GEMM block
#define GTHREADS 128  // 4 waves

typedef __attribute__((ext_vector_type(16))) _Float16 v16h;
typedef __attribute__((ext_vector_type(8)))  _Float16 v8h;
typedef __attribute__((ext_vector_type(8)))  float    v8f;

// ---------------- zero workspace (grid-stride, float4) ----------------------
__global__ void zero_f32(float* __restrict__ p, int n4) {
    int i = blockIdx.x * blockDim.x + threadIdx.x;
    int stride = gridDim.x * blockDim.x;
    float4* p4 = (float4*)p;
    for (; i < n4; i += stride) p4[i] = make_float4(0.f, 0.f, 0.f, 0.f);
}

// ---------------- f32 -> f16 weight conversion ------------------------------
__global__ void cvt_f16(const float* __restrict__ src, _Float16* __restrict__ dst, int n) {
    int i = blockIdx.x * blockDim.x + threadIdx.x;
    int stride = gridDim.x * blockDim.x;
    for (; i < n; i += stride) dst[i] = (_Float16)src[i];
}

// ---------------- edge scatter: agg[dst] += x[src] --------------------------
// One wave32 per edge; each lane moves a float4 (32*4 = 128 dims).
// Edge id forced uniform -> index loads become scalar s_load, row gather uses
// SADDR-form global_load_b128.
__global__ void scatter_agg(const float* __restrict__ X,
                            const int* __restrict__ src,
                            const int* __restrict__ dst,
                            float* __restrict__ agg, int nEdges) {
    const int lane = threadIdx.x & 31;
    // wave-uniform edge id, made SGPR-provable for the compiler
    int edge = __builtin_amdgcn_readfirstlane(
        blockIdx.x * (blockDim.x >> 5) + (threadIdx.x >> 5));
    if (edge >= nEdges) return;                 // uniform scalar branch
    const int s = src[edge];                    // s_load
    const int d = dst[edge];                    // s_load
    const float4 v = *(const float4*)(X + (size_t)s * KDIM + lane * 4);
    float* a = agg + (size_t)d * KDIM + lane * 4;
    unsafeAtomicAdd(a + 0, v.x);                // global_atomic_add_f32
    unsafeAtomicAdd(a + 1, v.y);
    unsafeAtomicAdd(a + 2, v.z);
    unsafeAtomicAdd(a + 3, v.w);
}

// ---------------- fused (X+AGG) @ W + b [, relu] via WMMA -------------------
// Block: 128 threads (4 waves), 64 node-rows. K=128 fully staged in LDS.
// Wave w owns row-tile w (16 rows) and loops over DOUT/16 col-tiles, unrolled
// x2 so two independent accumulator chains hide WMMA->WMMA RAW latency.
template <int DOUT, bool RELU>
__global__ void gin_gemm(const float* __restrict__ X,
                         const float* __restrict__ AGG,
                         const _Float16* __restrict__ Wh,   // [KDIM, DOUT] row-major
                         const float* __restrict__ bias,    // [DOUT]
                         float* __restrict__ Out,           // [N, DOUT]
                         int nNodes) {
    __shared__ __align__(16) _Float16 Ah[GROWS * LPAD];   // (x+agg) tile, f16
    __shared__ __align__(16) _Float16 Wt[DOUT * LPAD];    // W transposed: [n][k]

    const int tid = threadIdx.x;
    const int rowbase = blockIdx.x * GROWS;
    const bool fullTile = (rowbase + GROWS) <= nNodes;    // block-uniform

    // Stage A = (X + AGG) -> f16, row-major with padded stride.
    for (int i = tid; i < GROWS * KDIM; i += GTHREADS) {
        int r = i >> 7, c = i & (KDIM - 1);
        int node = rowbase + r;
        float v = 0.f;
        if (node < nNodes) {
            size_t o = (size_t)node * KDIM + c;
            v = X[o] + AGG[o];
        }
        Ah[r * LPAD + c] = (_Float16)v;
    }
    // Stage W transposed: Wt[n][k] = Wh[k][n] (coalesced global reads).
    constexpr int DSH = (DOUT == 128) ? 7 : 6;
    for (int i = tid; i < KDIM * DOUT; i += GTHREADS) {
        int k = i >> DSH, n = i & (DOUT - 1);
        Wt[n * LPAD + k] = Wh[i];
    }
    __syncthreads();

    const int wave = tid >> 5;
    const int lane = tid & 31;
    const int lh = lane >> 4;    // lane half: selects K sub-range per ISA layout
    const int lm = lane & 15;
    const int arow = wave * 16 + lm;          // A fragment row (M = lane%16)
    constexpr int NCT = DOUT >> 4;

    // Preload the four A fragments (loop-invariant across col-tiles).
    v16h afrag[4];
    #pragma unroll
    for (int kt = 0; kt < 4; ++kt) {
        const int kb = kt * 32;
        // A 16x32 f16 frag: elems 0..7 <- K=kb+8*lh.., elems 8..15 <- K=kb+16+8*lh..
        v8h alo = *(const v8h*)(Ah + arow * LPAD + kb + lh * 8);
        v8h ahi = *(const v8h*)(Ah + arow * LPAD + kb + 16 + lh * 8);
        afrag[kt] = __builtin_shufflevector(alo, ahi,
                    0,1,2,3,4,5,6,7,8,9,10,11,12,13,14,15);
    }

    #pragma unroll 2
    for (int ct = 0; ct < NCT; ++ct) {
        const int col = ct * 16 + lm;         // B/C fragment col (N = lane%16)
        v8f acc = {};
        #pragma unroll
        for (int kt = 0; kt < 4; ++kt) {
            const int kb = kt * 32;
            // B 32x16 f16 frag: elems 0..7 <- K=kb+8*lh.., elems 8..15 <- K=kb+16+8*lh..
            v8h blo = *(const v8h*)(Wt + col * LPAD + kb + lh * 8);
            v8h bhi = *(const v8h*)(Wt + col * LPAD + kb + 16 + lh * 8);
            v16h b = __builtin_shufflevector(blo, bhi,
                     0,1,2,3,4,5,6,7,8,9,10,11,12,13,14,15);
            acc = __builtin_amdgcn_wmma_f32_16x16x32_f16(
                      false, afrag[kt], false, b, (short)0, acc, false, false);
        }
        // C/D layout: VGPR v -> row (v + 8*lh), col = lane%16.
        const float bv = bias[col];
        const size_t obase = (size_t)(rowbase + wave * 16 + lh * 8) * DOUT + col;
        if (fullTile) {                       // uniform branch: no exec juggling
            #pragma unroll
            for (int v = 0; v < 8; ++v) {
                float o = acc[v] + bv;
                if (RELU) o = fmaxf(o, 0.f);
                Out[obase + (size_t)v * DOUT] = o;
            }
        } else {
            #pragma unroll
            for (int v = 0; v < 8; ++v) {
                int node = rowbase + wave * 16 + lh * 8 + v;
                if (node < nNodes) {
                    float o = acc[v] + bv;
                    if (RELU) o = fmaxf(o, 0.f);
                    Out[(size_t)node * DOUT + col] = o;
                }
            }
        }
    }
}

// ---------------------------------------------------------------------------
extern "C" void kernel_launch(void* const* d_in, const int* in_sizes, int n_in,
                              void* d_out, int out_size, void* d_ws, size_t ws_size,
                              hipStream_t stream) {
    const float* x  = (const float*)d_in[0];
    const int*   ei = (const int*)  d_in[1];   // [2, E]: row0=src, row1=dst
    const float* W1 = (const float*)d_in[2];
    const float* b1 = (const float*)d_in[3];
    const float* Wm = (const float*)d_in[4];
    const float* bm = (const float*)d_in[5];
    const float* W2 = (const float*)d_in[6];
    const float* b2 = (const float*)d_in[7];
    float* out = (float*)d_out;

    const int* src = ei;
    const int* dst = ei + N_EDGES;

    const size_t feat = (size_t)N_NODES * KDIM;     // 12.8M floats
    float* agg = (float*)d_ws;
    float* h1  = agg + feat;
    float* h2  = h1 + feat;
    _Float16* W1h = (_Float16*)(h2 + feat);
    _Float16* Wmh = W1h + KDIM * 128;
    _Float16* W2h = Wmh + KDIM * 128;

    // Weight conversion (tiny, once per launch).
    cvt_f16<<<64, 256, 0, stream>>>(W1, W1h, KDIM * 128);
    cvt_f16<<<64, 256, 0, stream>>>(Wm, Wmh, KDIM * 128);
    cvt_f16<<<32, 256, 0, stream>>>(W2, W2h, KDIM * 64);

    const int zeroBlocks = 1024;                     // grid-stride over 3.2M float4
    const int scatBlocks = (N_EDGES * 32 + 255) / 256;
    const int gemmBlocks = (N_NODES + GROWS - 1) / GROWS;

    // Layer 1: h1 = relu((x+agg) @ W1 + b1)
    zero_f32<<<zeroBlocks, 256, 0, stream>>>(agg, (int)(feat / 4));
    scatter_agg<<<scatBlocks, 256, 0, stream>>>(x, src, dst, agg, N_EDGES);
    gin_gemm<128, true><<<gemmBlocks, GTHREADS, 0, stream>>>(x, agg, W1h, b1, h1, N_NODES);

    // Layer 2: h2 = relu((h1+agg) @ Wm + bm)
    zero_f32<<<zeroBlocks, 256, 0, stream>>>(agg, (int)(feat / 4));
    scatter_agg<<<scatBlocks, 256, 0, stream>>>(h1, src, dst, agg, N_EDGES);
    gin_gemm<128, true><<<gemmBlocks, GTHREADS, 0, stream>>>(h1, agg, Wmh, bm, h2, N_NODES);

    // Layer 3: out = (h2+agg) @ W2 + b2
    zero_f32<<<zeroBlocks, 256, 0, stream>>>(agg, (int)(feat / 4));
    scatter_agg<<<scatBlocks, 256, 0, stream>>>(h2, src, dst, agg, N_EDGES);
    gin_gemm<64, false><<<gemmBlocks, GTHREADS, 0, stream>>>(h2, agg, W2h, b2, out, N_NODES);
}
